// Encoder_78417512891256
// MI455X (gfx1250) — matrix operand order repaired and checked by reading, not compiled, not run on hardware
//
#include <hip/hip_runtime.h>

// ---------------- problem constants (from reference setup_inputs) ----------
#define BB   8
#define NP   8192
#define MM   64
#define KQ   3
#define KN   (KQ * NP)      // 24576 columns per batch for the point MLP
#define SOMK 9
#define EPSV 1e-5f

typedef __attribute__((ext_vector_type(16))) _Float16 v16h;
typedef __attribute__((ext_vector_type(8)))  _Float16 v8h;
typedef __attribute__((ext_vector_type(8)))  float    v8f;
typedef __attribute__((ext_vector_type(4)))  unsigned int v4u;
typedef __attribute__((ext_vector_type(8)))  int      v8i;
typedef __attribute__((ext_vector_type(4)))  int      v4i;

// ---------------------------------------------------------------------------
// Utility kernels
// ---------------------------------------------------------------------------
__global__ void k_zero(float* __restrict__ p, int n) {
    int i = blockIdx.x * blockDim.x + threadIdx.x;
    if (i < n) p[i] = 0.f;
}

// fp32 weight [rows][cols_src] -> f16 [rows][cols_dst], zero-padded columns
__global__ void k_wconv(const float* __restrict__ src, _Float16* __restrict__ dst,
                        int rows, int cs, int cd) {
    int i = blockIdx.x * blockDim.x + threadIdx.x;
    if (i >= rows * cd) return;
    int r = i / cd, c = i % cd;
    dst[i] = (c < cs) ? (_Float16)src[r * cs + c] : (_Float16)0.f;
}

// ---------------------------------------------------------------------------
// SOM top-3 assignment + cluster sums (LDS-reduced atomics)
// ---------------------------------------------------------------------------
__global__ void k_assign(const float* __restrict__ x, const float* __restrict__ node,
                         int* __restrict__ minidx, float* __restrict__ cnt,
                         float* __restrict__ sums) {
    __shared__ float nx[MM], ny[MM], nz[MM];
    __shared__ float sc[MM], sx[MM], sy[MM], sz[MM];
    const int BPB = NP / 256;                 // blocks per batch
    int b = blockIdx.x / BPB;
    int p = (blockIdx.x % BPB) * 256 + threadIdx.x;
    int t = threadIdx.x;
    if (t < MM) {
        nx[t] = node[b * 3 * MM + 0 * MM + t];
        ny[t] = node[b * 3 * MM + 1 * MM + t];
        nz[t] = node[b * 3 * MM + 2 * MM + t];
        sc[t] = 0.f; sx[t] = 0.f; sy[t] = 0.f; sz[t] = 0.f;
    }
    __syncthreads();
    float px = x[b * 3 * NP + 0 * NP + p];
    float py = x[b * 3 * NP + 1 * NP + p];
    float pz = x[b * 3 * NP + 2 * NP + p];
    float d0 = 3.4e38f, d1 = 3.4e38f, d2 = 3.4e38f;
    int   i0 = 0, i1 = 0, i2 = 0;
    for (int m = 0; m < MM; ++m) {
        float dx = nx[m] - px, dy = ny[m] - py, dz = nz[m] - pz;
        float d = dx * dx + dy * dy + dz * dz;
        if (d < d0)      { d2 = d1; i2 = i1; d1 = d0; i1 = i0; d0 = d; i0 = m; }
        else if (d < d1) { d2 = d1; i2 = i1; d1 = d;  i1 = m; }
        else if (d < d2) { d2 = d;  i2 = m; }
    }
    // k-block-of-N layout matches reference reshape
    minidx[b * KN + 0 * NP + p] = i0;
    minidx[b * KN + 1 * NP + p] = i1;
    minidx[b * KN + 2 * NP + p] = i2;
    int idx[3] = { i0, i1, i2 };
    for (int q = 0; q < 3; ++q) {
        atomicAdd(&sc[idx[q]], 1.f);
        atomicAdd(&sx[idx[q]], px);
        atomicAdd(&sy[idx[q]], py);
        atomicAdd(&sz[idx[q]], pz);
    }
    __syncthreads();
    if (t < MM) {
        atomicAdd(&cnt[b * MM + t], sc[t]);
        atomicAdd(&sums[b * 3 * MM + 0 * MM + t], sx[t]);
        atomicAdd(&sums[b * 3 * MM + 1 * MM + t], sy[t]);
        atomicAdd(&sums[b * 3 * MM + 2 * MM + t], sz[t]);
    }
}

__global__ void k_mean(const float* __restrict__ sums, const float* __restrict__ cnt,
                       float* __restrict__ mean) {
    int i = blockIdx.x * blockDim.x + threadIdx.x;   // B*M
    if (i >= BB * MM) return;
    int b = i / MM, m = i % MM;
    float inv = 1.f / (cnt[b * MM + m] + EPSV);
    for (int c = 0; c < 3; ++c)
        mean[b * 3 * MM + c * MM + m] = sums[b * 3 * MM + c * MM + m] * inv;
}

// build f16 x_aug padded to 32 rows: [B][32][kN]
__global__ void k_aug(const float* __restrict__ x, const float* __restrict__ sn,
                      const float* __restrict__ mean, const int* __restrict__ minidx,
                      _Float16* __restrict__ actA) {
    long long i = (long long)blockIdx.x * blockDim.x + threadIdx.x;   // B*kN
    if (i >= (long long)BB * KN) return;
    int b = (int)(i / KN);
    int n = (int)(i % KN);
    int p = n % NP;
    int idx = minidx[b * KN + n];
    _Float16* dst = actA + (long long)b * 32 * KN + n;
    for (int c = 0; c < 3; ++c)
        dst[(long long)c * KN] =
            (_Float16)(x[b * 3 * NP + c * NP + p] - mean[b * 3 * MM + c * MM + idx]);
    for (int c = 0; c < 3; ++c)
        dst[(long long)(3 + c) * KN] = (_Float16)sn[b * 3 * NP + c * NP + p];
    for (int c = 6; c < 32; ++c)
        dst[(long long)c * KN] = (_Float16)0.f;
}

// ---------------------------------------------------------------------------
// WMMA GEMM with TDM-staged B stripe:
//   out[Cout][ncols] = relu(W[Cout][Cinp] * act + bias)
// One block owns one 16-column stripe of one batch. Wave 0 issues a single
// TENSOR_LOAD_TO_LDS (D# per ISA 8.3/8.4) pulling the whole Cinp x 16 f16
// stripe into LDS, waits TENSORcnt, then all 8 waves sweep the row tiles
// reading B fragments from LDS (lane L = column L%16, rows k0+8h+0..7 and k0+16+8h+0..7, h=L/16).
// MODE 0: store f16 activations   MODE 1: atomic segment-max into node feats
// MODE 2: atomic max into d_out (global max-pool fused)
// ---------------------------------------------------------------------------
template <int MODE>
__global__ void gemm_wmma(const _Float16* __restrict__ A, int lda,
                          const _Float16* __restrict__ Bact, long long bstr, int ldb,
                          const float* __restrict__ bias,
                          _Float16* __restrict__ Cact, long long cstr, int ldo,
                          float* __restrict__ outF, const int* __restrict__ minidx,
                          float* __restrict__ col0,
                          int Cout, int Cinp, int ncols) {
    __shared__ __align__(64) _Float16 smem[768 * 16];   // max Cinp = 768 -> 24 KB
    int colTiles = ncols >> 4;
    int rowTiles = Cout >> 4;
    int b    = blockIdx.x / colTiles;
    int n0   = (blockIdx.x % colTiles) << 4;
    int lane = threadIdx.x & 31;
    int wid  = threadIdx.x >> 5;

    const _Float16* Bb = Bact + (long long)b * bstr + n0;

    if (wid == 0) {
        // ---- Tensor DMA descriptor (D#), groups 0/1; 2D tile, 2-byte elems ----
        unsigned long long ga = (unsigned long long)(const void*)Bb;
        unsigned lds = (unsigned)(unsigned long long)(const void*)&smem[0];
        v4u g0; v8i g1;
        v4i g2 = {0, 0, 0, 0};
        v4i g3 = {0, 0, 0, 0};
        v8i g4 = {0, 0, 0, 0, 0, 0, 0, 0};
        g0[0] = 1u;                                            // count=1, user mode
        g0[1] = lds;                                           // lds_addr (bytes)
        g0[2] = (unsigned)(ga & 0xFFFFFFFFull);                // global_addr[31:0]
        g0[3] = (unsigned)((ga >> 32) & 0x01FFFFFFull)         // global_addr[56:32]
                | 0x80000000u;                                 // type=2 ("image")
        g1[0] = (int)(1u << 16);                               // data_size=1 (2B)
        g1[1] = (int)(((unsigned)ldb & 0xFFFFu) << 16);        // tensor_dim0[15:0]
        g1[2] = (int)((((unsigned)ldb >> 16) & 0xFFFFu)        // tensor_dim0[31:16]
                | (((unsigned)Cinp & 0xFFFFu) << 16));         // tensor_dim1[15:0]
        g1[3] = (int)((((unsigned)Cinp >> 16) & 0xFFFFu)       // tensor_dim1[31:16]
                | (16u << 16));                                // tile_dim0 = 16
        g1[4] = (int)((unsigned)Cinp & 0xFFFFu);               // tile_dim1 = Cinp rows
        g1[5] = (int)(unsigned)ldb;                            // tensor_dim0_stride lo32
        g1[6] = 0;                                             // stride hi / dim1_stride
        g1[7] = 0;
        __builtin_amdgcn_tensor_load_to_lds(g0, g1, g2, g3, g4, 0);
        __builtin_amdgcn_s_wait_tensorcnt(0);
    }
    __syncthreads();

    // A fragment addressing (ISA 7.12.2, 16-bit A 16x32)
    int mrowOff = lane & 15;
    int kSel    = (lane >> 4) << 3;
    int Nc      = lane & 15;
    int mAdd    = (lane >> 4) << 3;

    for (int rt = wid; rt < rowTiles; rt += 8) {
        int m0 = rt << 4;
        const _Float16* Arow = A + (long long)(m0 + mrowOff) * lda + kSel;
        v8f acc = {};
        for (int k0 = 0; k0 < Cinp; k0 += 32) {
            v8h alo = *(const v8h*)(Arow + k0);        // K = base+0..7
            v8h ahi = *(const v8h*)(Arow + k0 + 16);   // K = base+16..23
            v16h a;
            #pragma unroll
            for (int j = 0; j < 8; ++j) { a[j] = alo[j]; a[j + 8] = ahi[j]; }
            // B fragment from LDS: lane column N=lane%16, K = base+0..7 and base+16..23 (same as A)
            v16h bf;
            #pragma unroll
            for (int j = 0; j < 8; ++j) {
                bf[j]     = smem[((k0 + kSel + j) << 4) + Nc];
                bf[j + 8] = smem[((k0 + 16 + kSel + j) << 4) + Nc];
            }
            acc = __builtin_amdgcn_wmma_f32_16x16x32_f16(
                false, a, false, bf, (short)0, acc, false, false);
        }
        #pragma unroll
        for (int r = 0; r < 8; ++r) {
            int row = m0 + r + mAdd;
            float v = acc[r] + bias[row];
            v = v > 0.f ? v : 0.f;
            int n = n0 + Nc;
            if (MODE == 0) {
                Cact[(long long)b * cstr + (long long)row * ldo + n] = (_Float16)v;
            } else if (MODE == 1) {
                int node = minidx[b * ncols + n];
                // post-ReLU values >= 0: float bits compare as ints
                atomicMax((int*)&outF[((long long)b * Cout + row) * MM + node],
                          __float_as_int(v));
                if (n == 0) col0[b * Cout + row] = v;  // first[:, :, 0] for empty nodes
            } else {
                atomicMax((int*)&outF[(long long)b * Cout + row], __float_as_int(v));
            }
        }
    }
}

// empty nodes take first[:, :, 0]
__global__ void k_fix(float* __restrict__ node_max, const float* __restrict__ col0,
                      const float* __restrict__ cnt) {
    int i = blockIdx.x * blockDim.x + threadIdx.x;   // B*384*M
    if (i >= BB * 384 * MM) return;
    int m = i % MM;
    int c = (i / MM) % 384;
    int b = i / (384 * MM);
    if (cnt[b * MM + m] <= 0.f) node_max[i] = col0[b * 384 + c];
}

// center = mean over 9 SOM neighbors of cluster_mean
__global__ void k_center(const float* __restrict__ mean, const int* __restrict__ knnI,
                         float* __restrict__ center) {
    int i = blockIdx.x * blockDim.x + threadIdx.x;   // B*M
    if (i >= BB * MM) return;
    int b = i / MM, m = i % MM;
    float s[3] = { 0.f, 0.f, 0.f };
    for (int kk = 0; kk < SOMK; ++kk) {
        int nb = knnI[b * MM * SOMK + m * SOMK + kk];
        for (int c = 0; c < 3; ++c) s[c] += mean[b * 3 * MM + c * MM + nb];
    }
    for (int c = 0; c < 3; ++c)
        center[b * 3 * MM + c * MM + m] = s[c] * (1.f / SOMK);
}

// KNN-module input: [B][416][576] f16 (rows 0-2 centered coords, 3-386 feats, pad 0)
__global__ void k_knnaug(const float* __restrict__ mean, const float* __restrict__ center,
                         const float* __restrict__ node_max, const int* __restrict__ knnI,
                         _Float16* __restrict__ aug) {
    int b   = blockIdx.x / (MM * SOMK);
    int col = blockIdx.x % (MM * SOMK);
    int m = col / SOMK, kk = col % SOMK;
    int nb = knnI[b * MM * SOMK + m * SOMK + kk];
    _Float16* dst = aug + (long long)b * 416 * 576 + col;
    for (int row = threadIdx.x; row < 416; row += blockDim.x) {
        float v;
        if (row < 3)
            v = mean[b * 3 * MM + row * MM + nb] - center[b * 3 * MM + row * MM + m];
        else if (row < 387)
            v = node_max[((long long)b * 384 + (row - 3)) * MM + nb];
        else
            v = 0.f;
        dst[(long long)row * 576] = (_Float16)v;
    }
}

// final-MLP input: [B][544][64] f16 = [center(3); max_k knn_g(512); pad(29)]
__global__ void k_finin(const float* __restrict__ center, const _Float16* __restrict__ g,
                        _Float16* __restrict__ fin) {
    int i = blockIdx.x * blockDim.x + threadIdx.x;   // B*M
    if (i >= BB * MM) return;
    int b = i / MM, m = i % MM;
    _Float16* dst = fin + (long long)b * 544 * 64 + m;
    for (int c = 0; c < 3; ++c)
        dst[(long long)c * 64] = (_Float16)center[b * 3 * MM + c * MM + m];
    for (int c = 0; c < 512; ++c) {
        const _Float16* src = g + ((long long)b * 512 + c) * 576 + m * SOMK;
        float mx = (float)src[0];
        for (int kk = 1; kk < SOMK; ++kk) {
            float v = (float)src[kk];
            mx = v > mx ? v : mx;
        }
        dst[(long long)(3 + c) * 64] = (_Float16)mx;
    }
    for (int c = 515; c < 544; ++c) dst[(long long)c * 64] = (_Float16)0.f;
}

// ---------------------------------------------------------------------------
extern "C" void kernel_launch(void* const* d_in, const int* in_sizes, int n_in,
                              void* d_out, int out_size, void* d_ws, size_t ws_size,
                              hipStream_t stream) {
    const float* x    = (const float*)d_in[0];
    const float* sn   = (const float*)d_in[1];
    const float* node = (const float*)d_in[2];
    const int*   knnI = (const int*)d_in[3];
    const float* pr_w0 = (const float*)d_in[4];  const float* pr_b0 = (const float*)d_in[5];
    const float* pr_w1 = (const float*)d_in[6];  const float* pr_b1 = (const float*)d_in[7];
    const float* pr_w2 = (const float*)d_in[8];  const float* pr_b2 = (const float*)d_in[9];
    const float* pr_w3 = (const float*)d_in[10]; const float* pr_b3 = (const float*)d_in[11];
    const float* kw0   = (const float*)d_in[12]; const float* kb0   = (const float*)d_in[13];
    const float* kw1   = (const float*)d_in[14]; const float* kb1   = (const float*)d_in[15];
    const float* fw0   = (const float*)d_in[16]; const float* fb0   = (const float*)d_in[17];
    const float* fw1   = (const float*)d_in[18]; const float* fb1   = (const float*)d_in[19];
    float* out = (float*)d_out;

    // -------- workspace carve-out (256B aligned) --------
    char* ws = (char*)d_ws;
    size_t off = 0;
    auto alloc = [&](size_t bytes) -> void* {
        void* p = ws + off;
        off = (off + bytes + 255) & ~(size_t)255;
        return p;
    };
    float* cnt      = (float*)alloc(BB * MM * 4);
    float* sums     = (float*)alloc(BB * 3 * MM * 4);
    float* mean     = (float*)alloc(BB * 3 * MM * 4);
    float* centerB  = (float*)alloc(BB * 3 * MM * 4);
    int*   minidx   = (int*)alloc((size_t)BB * KN * 4);
    float* node_max = (float*)alloc((size_t)BB * 384 * MM * 4);
    float* col0     = (float*)alloc(BB * 384 * 4);
    _Float16* wA0 = (_Float16*)alloc(64  * 32  * 2);
    _Float16* wA1 = (_Float16*)alloc(128 * 64  * 2);
    _Float16* wA2 = (_Float16*)alloc(256 * 128 * 2);
    _Float16* wA3 = (_Float16*)alloc(384 * 320 * 2);
    _Float16* wK0 = (_Float16*)alloc(512 * 416 * 2);
    _Float16* wK1 = (_Float16*)alloc(512 * 512 * 2);
    _Float16* wF0 = (_Float16*)alloc(768 * 544 * 2);
    _Float16* wF1 = (_Float16*)alloc(1024 * 768 * 2);
    _Float16* actA   = (_Float16*)alloc((size_t)BB * 32  * KN * 2);
    _Float16* actCat = (_Float16*)alloc((size_t)BB * 320 * KN * 2); // rows0-63 h0, 64-319 h2
    _Float16* actH1  = (_Float16*)alloc((size_t)BB * 128 * KN * 2);
    _Float16* knnAug = (_Float16*)alloc((size_t)BB * 416 * 576 * 2);
    _Float16* knnG0  = (_Float16*)alloc((size_t)BB * 512 * 576 * 2);
    _Float16* knnG1  = (_Float16*)alloc((size_t)BB * 512 * 576 * 2);
    _Float16* finIn  = (_Float16*)alloc((size_t)BB * 544 * 64 * 2);
    _Float16* finH   = (_Float16*)alloc((size_t)BB * 768 * 64 * 2);
    (void)ws_size; (void)n_in; (void)in_sizes; (void)out_size;

    // -------- reset atomic accumulators / output every call --------
    k_zero<<<(BB * MM + 255) / 256, 256, 0, stream>>>(cnt, BB * MM);
    k_zero<<<(BB * 3 * MM + 255) / 256, 256, 0, stream>>>(sums, BB * 3 * MM);
    k_zero<<<(BB * 384 * MM + 255) / 256, 256, 0, stream>>>(node_max, BB * 384 * MM);
    k_zero<<<(BB * 1024 + 255) / 256, 256, 0, stream>>>(out, BB * 1024);

    // -------- weights fp32 -> f16 (with K padding to x32) --------
    auto wc = [&](const float* s, _Float16* d, int r, int cs, int cd) {
        k_wconv<<<(r * cd + 255) / 256, 256, 0, stream>>>(s, d, r, cs, cd);
    };
    wc(pr_w0, wA0, 64, 6, 32);
    wc(pr_w1, wA1, 128, 64, 64);
    wc(pr_w2, wA2, 256, 128, 128);
    wc(pr_w3, wA3, 384, 320, 320);
    wc(kw0,   wK0, 512, 387, 416);
    wc(kw1,   wK1, 512, 512, 512);
    wc(fw0,   wF0, 768, 515, 544);
    wc(fw1,   wF1, 1024, 768, 768);

    // -------- SOM assignment, means, augmented input --------
    k_assign<<<BB * (NP / 256), 256, 0, stream>>>(x, node, minidx, cnt, sums);
    k_mean<<<(BB * MM + 255) / 256, 256, 0, stream>>>(sums, cnt, mean);
    k_aug<<<(int)(((long long)BB * KN + 255) / 256), 256, 0, stream>>>(x, sn, mean, minidx, actA);

    auto gemm = [&](int mode, const _Float16* A, int lda, const _Float16* Ba,
                    long long bstr, int ldb, const float* bias, _Float16* C,
                    long long cstr, int ldo, float* oF, const int* mi, float* c0,
                    int Cout, int Cinp, int ncols) {
        int blocks = (ncols / 16) * BB;   // one block per 16-col stripe per batch
        if (mode == 0)
            gemm_wmma<0><<<blocks, 256, 0, stream>>>(A, lda, Ba, bstr, ldb, bias, C,
                                                     cstr, ldo, oF, mi, c0, Cout, Cinp, ncols);
        else if (mode == 1)
            gemm_wmma<1><<<blocks, 256, 0, stream>>>(A, lda, Ba, bstr, ldb, bias, C,
                                                     cstr, ldo, oF, mi, c0, Cout, Cinp, ncols);
        else
            gemm_wmma<2><<<blocks, 256, 0, stream>>>(A, lda, Ba, bstr, ldb, bias, C,
                                                     cstr, ldo, oF, mi, c0, Cout, Cinp, ncols);
    };

    // -------- PointResNet (f16 WMMA chain over kN columns) --------
    gemm(0, wA0, 32,  actA,   (long long)32 * KN,  KN, pr_b0,
         actCat, (long long)320 * KN, KN, nullptr, nullptr, nullptr, 64, 32, KN);
    gemm(0, wA1, 64,  actCat, (long long)320 * KN, KN, pr_b1,
         actH1,  (long long)128 * KN, KN, nullptr, nullptr, nullptr, 128, 64, KN);
    gemm(0, wA2, 128, actH1,  (long long)128 * KN, KN, pr_b2,
         actCat + (long long)64 * KN, (long long)320 * KN, KN,
         nullptr, nullptr, nullptr, 256, 128, KN);
    // last layer fused with segment-max into node features (B,384,kN never stored)
    gemm(1, wA3, 320, actCat, (long long)320 * KN, KN, pr_b3,
         nullptr, 0, 0, node_max, minidx, col0, 384, 320, KN);
    k_fix<<<(BB * 384 * MM + 255) / 256, 256, 0, stream>>>(node_max, col0, cnt);

    // -------- KNN module over SOM nodes --------
    k_center<<<(BB * MM + 255) / 256, 256, 0, stream>>>(mean, knnI, centerB);
    k_knnaug<<<BB * MM * SOMK, 256, 0, stream>>>(mean, centerB, node_max, knnI, knnAug);
    gemm(0, wK0, 416, knnAug, (long long)416 * 576, 576, kb0,
         knnG0, (long long)512 * 576, 576, nullptr, nullptr, nullptr, 512, 416, 576);
    gemm(0, wK1, 512, knnG0,  (long long)512 * 576, 576, kb1,
         knnG1, (long long)512 * 576, 576, nullptr, nullptr, nullptr, 512, 512, 576);

    // -------- final PointNet + fused global max pool --------
    k_finin<<<(BB * MM + 63) / 64, 64, 0, stream>>>(centerB, knnG1, finIn);
    gemm(0, wF0, 544, finIn, (long long)544 * 64, 64, fb0,
         finH, (long long)768 * 64, 64, nullptr, nullptr, nullptr, 768, 544, 64);
    gemm(2, wF1, 768, finH,  (long long)768 * 64, 64, fb1,
         nullptr, 0, 0, out, nullptr, nullptr, 1024, 768, 64);
}
